// unit_agcn_39668317946133
// MI455X (gfx1250) — compile-verified, matrix operand order checked by reading, not run on hardware
//
#include <hip/hip_runtime.h>
#include <hip/hip_bf16.h>
#include <math.h>

// ---------------------------------------------------------------------------
// Fused unit-AGCN block for MI455X (gfx1250, wave32, WMMA 16x16x32 f16)
// dims: N=128, C=64, T=128, V=25, S=3, rel=8, inner=24
// ---------------------------------------------------------------------------

// Flip to 0 if the assembler rejects the async-to-LDS mnemonic.
#define AGCN_ASYNC 1

typedef __attribute__((ext_vector_type(16))) _Float16 v16h;
typedef __attribute__((ext_vector_type(8)))  float    v8f;

#define NB   128
#define CC   64
#define TT   128
#define VV   25
#define SS   3
#define REL  8
#define QKN  48          // 2 * inner
#define KCAT 192         // S * C
#define VP   32          // V padded to WMMA M

// padded LDS row strides (halfs / floats) -> conflict-free b128 lane patterns
#define LDY  72          // ynh:  144 B rows, start bank r*36 mod 64
#define LDX  40          // xvT:   80 B rows, start bank r*20 mod 64
#define LDA  40          // Ah:    80 B rows
#define LDZ  200         // zcat: 400 B rows, start bank r*36 mod 64
#define LDQ  52          // qkf (f32): 208 B rows

union frag16_u { v16h h; uint32_t u[8]; };

// Load a 16x32 (f16) A/B fragment from a row-major tile (LDS or global).
// ISA 7.12.2 16-bit A layout: lane = row mod 16; lanes 0-15 hold K pairs
// {0,2,4,6,16,18,20,22}+{0,1}; lanes 16-31 the same +8.
__device__ __forceinline__ v16h load_frag16(const _Float16* tile, int ld, int lane) {
    const int r    = lane & 15;
    const int koff = (lane & 16) >> 1;          // +8 for upper half-wave
    const _Float16* row = tile + r * ld + koff;
    frag16_u f;
#pragma unroll
    for (int p = 0; p < 8; ++p) {
        const int k = (p < 4) ? (2 * p) : (8 + 2 * p);   // 0,2,4,6,16,18,20,22
        f.u[p] = *(const uint32_t*)(row + k);            // aligned: ld even, k even
    }
    return f.h;
}

__device__ __forceinline__ v8f wmma_f16(v16h a, v16h b, v8f c) {
    return __builtin_amdgcn_wmma_f32_16x16x32_f16(false, a, false, b,
                                                  (short)0, c, false, false);
}

// ---------------------------------------------------------------------------
// Prep kernel: f32->f16 weight conversion + layout permute + BN folding.
// ---------------------------------------------------------------------------
__global__ void agcn_prep(const float* __restrict__ Wqk,     // (48,64)
                          const float* __restrict__ conv_w,  // (3,64,64) [h][o][c]
                          const float* __restrict__ conv_b,  // (3,64)
                          const float* __restrict__ bn_g, const float* __restrict__ bn_b,
                          const float* __restrict__ bn_m, const float* __restrict__ bn_v,
                          _Float16* __restrict__ wqk_h,      // (48,64)  N x K
                          _Float16* __restrict__ wcat_h,     // (64,192) N x K : [o][h*64+c]
                          float* __restrict__ bnscale, float* __restrict__ bnshift) {
    const int tid    = blockIdx.x * blockDim.x + threadIdx.x;
    const int stride = gridDim.x * blockDim.x;
    for (int i = tid; i < QKN * CC; i += stride)
        wqk_h[i] = (_Float16)Wqk[i];
    for (int i = tid; i < CC * KCAT; i += stride) {
        const int o = i / KCAT, k = i % KCAT, h = k >> 6, c = k & 63;
        wcat_h[i] = (_Float16)conv_w[(h * CC + o) * CC + c];
    }
    for (int o = tid; o < CC; o += stride) {
        const float s  = bn_g[o] * rsqrtf(bn_v[o] + 1e-5f);
        const float cb = conv_b[o] + conv_b[CC + o] + conv_b[2 * CC + o];
        bnscale[o] = s;
        bnshift[o] = (cb - bn_m[o]) * s + bn_b[o];
    }
}

// ---------------------------------------------------------------------------
// Main fused kernel: one block per (n, t) slice; 128 threads = 4 waves.
// ---------------------------------------------------------------------------
__global__ __launch_bounds__(128) void agcn_main(
        const float* __restrict__ x,        // (N,C,T,V)
        const float* __restrict__ PA,       // (S,V,V)
        const float* __restrict__ ln_g, const float* __restrict__ ln_b,
        const float* __restrict__ bqk,      // (48,)
        const _Float16* __restrict__ wqk_h, // (48,64)
        const _Float16* __restrict__ wcat_h,// (64,192)
        const float* __restrict__ bnscale, const float* __restrict__ bnshift,
        float* __restrict__ out)            // (N,C,T,V)
{
    __shared__ __align__(16) float    xtile[VV * CC];      // [v][c] f32 (LN src)
    __shared__ __align__(16) _Float16 ynh[VP * LDY];       // [v][c] LN output, rows 25..31 = 0
    __shared__ __align__(16) _Float16 xvT[CC * LDX];       // [c][j] x transposed, cols 25..31 = 0
    __shared__ __align__(16) float    qkf[VP * LDQ];       // qk projection (f32)
    __shared__ __align__(16) _Float16 Ah[SS * VP * LDA];   // per-head softmax(qk)*PA, zero-padded
    __shared__ __align__(16) _Float16 zcat[VP * LDZ];      // [v][h*64+c] aggregated messages

    const int tid  = threadIdx.x;
    const int lane = tid & 31;
    const int wave = tid >> 5;
    const int bid  = blockIdx.x;
    const int n    = bid >> 7;          // bid / T
    const int tt   = bid & 127;         // bid % T

    const float* xb = x + (size_t)n * CC * TT * VV + (size_t)tt * VV;   // + c*T*V + v

    // warm the weight tables through the WGP cache path (global_prefetch_b8)
    __builtin_prefetch(wcat_h + (tid << 6), 0, 0);
    if (tid < 48) __builtin_prefetch(wqk_h + (tid << 6), 0, 0);

    // ---- phase 0: zero pads + pull x tile into LDS ----
    for (int i = tid; i < SS * VP * LDA; i += 128) Ah[i]  = (_Float16)0.0f;
    for (int i = tid; i < CC * LDX;      i += 128) xvT[i] = (_Float16)0.0f;
    for (int i = tid; i < (VP - VV) * LDY; i += 128) ynh[VV * LDY + i] = (_Float16)0.0f;

#if AGCN_ASYNC
    // async global -> LDS copy (ASYNCcnt path); LDS offset = low 32 bits of
    // the flat pointer.
    for (int i = tid; i < VV * CC; i += 128) {
        const int v = i >> 6, c = i & 63;
        const uint32_t ldsa = (uint32_t)(uintptr_t)(xtile + i);
        const uint64_t ga   = (uint64_t)(uintptr_t)(xb + (size_t)c * TT * VV + v);
        asm volatile("global_load_async_to_lds_b32 %0, %1, off"
                     :: "v"(ldsa), "v"(ga) : "memory");
    }
    asm volatile("s_wait_asynccnt 0x0" ::: "memory");
    __syncthreads();
    // build the f16 transpose from LDS
    for (int i = tid; i < VV * CC; i += 128) {
        const int v = i >> 6, c = i & 63;
        xvT[c * LDX + v] = (_Float16)xtile[i];
    }
#else
    for (int i = tid; i < VV * CC; i += 128) {
        const int v = i >> 6, c = i & 63;
        const float val = xb[(size_t)c * TT * VV + v];
        xtile[i]         = val;
        xvT[c * LDX + v] = (_Float16)val;
    }
#endif

    // ---- LayerNorm over C per joint (25 rows, one thread each) ----
#if !AGCN_ASYNC
    __syncthreads();
#endif
    if (tid < VV) {
        const float* row = xtile + tid * CC;
        float mu = 0.0f;
#pragma unroll
        for (int c = 0; c < CC; ++c) mu += row[c];
        mu *= (1.0f / CC);
        float var = 0.0f;
#pragma unroll
        for (int c = 0; c < CC; ++c) { const float d = row[c] - mu; var += d * d; }
        const float inv = rsqrtf(var * (1.0f / CC) + 1e-5f);
#pragma unroll
        for (int c = 0; c < CC; ++c)
            ynh[tid * LDY + c] = (_Float16)((row[c] - mu) * inv * ln_g[c] + ln_b[c]);
    }
    __syncthreads();

    // ---- qk GEMM: (32x64) @ (64x48), 2x3 tiles, K in 2 steps ----
    for (int tile = wave; tile < 6; tile += 4) {
        const int mt = tile / 3, nt = tile % 3;
        v8f acc = {0, 0, 0, 0, 0, 0, 0, 0};
#pragma unroll
        for (int ks = 0; ks < 2; ++ks) {
            const v16h a = load_frag16(ynh   + mt * 16 * LDY + ks * 32, LDY, lane);
            const v16h b = load_frag16(wqk_h + nt * 16 * CC  + ks * 32, CC,  lane);
            acc = wmma_f16(a, b, acc);
        }
        const int col   = nt * 16 + (lane & 15);
        const int rbase = mt * 16 + ((lane & 16) >> 1);
        const float bias = bqk[col];
#pragma unroll
        for (int r = 0; r < 8; ++r)
            qkf[(rbase + r) * LDQ + col] = acc[r] + bias;
    }
    __syncthreads();

    // ---- per-head dots (K=8), softmax over j, multiply PA -> Ah (f16) ----
    if (tid < SS * VV) {
        const int h = tid / VV, i = tid % VV;
        const float* qrow = qkf + i * LDQ + h * REL;
        float d[VV];
        float mx = -1e30f;
#pragma unroll
        for (int j = 0; j < VV; ++j) {
            const float* krow = qkf + j * LDQ + (QKN / 2) + h * REL;
            float s = 0.0f;
#pragma unroll
            for (int dd = 0; dd < REL; ++dd) s += qrow[dd] * krow[dd];
            s *= 0.35355339059f;          // rel^-0.5
            d[j] = s;
            mx = fmaxf(mx, s);
        }
        float sum = 0.0f;
#pragma unroll
        for (int j = 0; j < VV; ++j) { const float e = __expf(d[j] - mx); d[j] = e; sum += e; }
        const float inv = 1.0f / sum;
        const float* pab = PA + (h * VV + i) * VV;
        _Float16* arow = Ah + h * VP * LDA + i * LDA;
#pragma unroll
        for (int j = 0; j < VV; ++j)
            arow[j] = (_Float16)(d[j] * inv * pab[j]);
    }
    __syncthreads();

    // ---- z GEMM per head: A(32x32pad) @ xvT-as-B (K=32pad, N=64); 24 tiles ----
    for (int tile = wave; tile < SS * 8; tile += 4) {
        const int h = tile >> 3, rem = tile & 7, mt = rem >> 2, nt = rem & 3;
        const v16h a = load_frag16(Ah  + h * VP * LDA + mt * 16 * LDA, LDA, lane);
        const v16h b = load_frag16(xvT + nt * 16 * LDX,                LDX, lane);
        v8f acc = {0, 0, 0, 0, 0, 0, 0, 0};
        acc = wmma_f16(a, b, acc);
        const int col   = h * CC + nt * 16 + (lane & 15);
        const int rbase = mt * 16 + ((lane & 16) >> 1);
#pragma unroll
        for (int r = 0; r < 8; ++r)
            zcat[(rbase + r) * LDZ + col] = (_Float16)acc[r];
    }
    __syncthreads();

    // ---- out GEMM: (32x192) @ (192x64), 2x4 tiles, K in 6 steps; fused epilogue ----
    for (int tile = wave; tile < 8; tile += 4) {
        const int mt = tile >> 2, nt = tile & 3;
        v8f acc = {0, 0, 0, 0, 0, 0, 0, 0};
#pragma unroll
        for (int ks = 0; ks < 6; ++ks) {
            const v16h a = load_frag16(zcat   + mt * 16 * LDZ  + ks * 32, LDZ,  lane);
            const v16h b = load_frag16(wcat_h + nt * 16 * KCAT + ks * 32, KCAT, lane);
            acc = wmma_f16(a, b, acc);
        }
        const int o     = nt * 16 + (lane & 15);
        const int rbase = mt * 16 + ((lane & 16) >> 1);
        const float sc = bnscale[o], sh = bnshift[o];
        float* ob = out + (size_t)n * CC * TT * VV + (size_t)o * TT * VV + (size_t)tt * VV;
#pragma unroll
        for (int r = 0; r < 8; ++r) {
            const int v = rbase + r;
            if (v < VV) {
                const float res = xb[(size_t)o * TT * VV + v];
                ob[v] = fmaxf(acc[r] * sc + sh + res, 0.0f);
            }
        }
    }
}

// ---------------------------------------------------------------------------
extern "C" void kernel_launch(void* const* d_in, const int* in_sizes, int n_in,
                              void* d_out, int out_size, void* d_ws, size_t ws_size,
                              hipStream_t stream) {
    (void)in_sizes; (void)n_in; (void)out_size; (void)ws_size;
    const float* x       = (const float*)d_in[0];
    const float* PA      = (const float*)d_in[1];
    const float* ln_g    = (const float*)d_in[2];
    const float* ln_b    = (const float*)d_in[3];
    const float* Wqk     = (const float*)d_in[4];
    const float* bqk     = (const float*)d_in[5];
    const float* conv_w  = (const float*)d_in[6];
    const float* conv_b  = (const float*)d_in[7];
    const float* bn_g    = (const float*)d_in[8];
    const float* bn_b    = (const float*)d_in[9];
    const float* bn_m    = (const float*)d_in[10];
    const float* bn_v    = (const float*)d_in[11];
    float* out = (float*)d_out;

    // workspace layout
    char* ws = (char*)d_ws;
    _Float16* wqk_h   = (_Float16*)(ws);                       //  6144 B (48*64)
    _Float16* wcat_h  = (_Float16*)(ws + 6144);                // 24576 B (64*192)
    float*    bnscale = (float*)   (ws + 6144 + 24576);        //   256 B
    float*    bnshift = (float*)   (ws + 6144 + 24576 + 256);  //   256 B

    agcn_prep<<<48, 256, 0, stream>>>(Wqk, conv_w, conv_b, bn_g, bn_b, bn_m, bn_v,
                                      wqk_h, wcat_h, bnscale, bnshift);

    agcn_main<<<NB * TT, 128, 0, stream>>>(x, PA, ln_g, ln_b, bqk,
                                           wqk_h, wcat_h, bnscale, bnshift, out);
}
